// PositionFeaturizer_76708115906773
// MI455X (gfx1250) — compile-verified, matrix operand order checked by reading, not run on hardware
//
#include <hip/hip_runtime.h>
#include <hip/hip_bf16.h>

#define N_NODES 10000
#define N_SRC   10000
#define N_EDGE  200000
#define HEADS   8
#define D_EMB   512
#define D_HID   1024
#define FEAT    32
#define D_H     544   // D_EMB + FEAT (logical)
#define D_HP    576   // padded to a multiple of BK=64
#define EPS_LN  1e-5f

typedef __attribute__((ext_vector_type(16))) _Float16 v16h;
typedef __attribute__((ext_vector_type(8)))  _Float16 v8h;
typedef __attribute__((ext_vector_type(8)))  float    v8f;

// ---------------------------------------------------------------------------
// CDNA5 async global->LDS path (ASYNCcnt). Probe-confirmed signature:
//   void __builtin_amdgcn_global_load_async_to_lds_b128(
//       int4(vector_size) AS1*, int4(vector_size) AS3*, imm offset, imm cpol)
// ---------------------------------------------------------------------------
#if defined(__gfx1250__) && __has_builtin(__builtin_amdgcn_global_load_async_to_lds_b128)
#define HAVE_ASYNC 1
typedef int v4i_vs __attribute__((vector_size(16)));
typedef __attribute__((address_space(1))) v4i_vs as1_v4i;
typedef __attribute__((address_space(3))) v4i_vs as3_v4i;
#else
#define HAVE_ASYNC 0
#endif

// copy 16 bytes global -> LDS (per-lane addresses)
__device__ inline void cp16(const _Float16* __restrict__ g, _Float16* l) {
#if HAVE_ASYNC
  // generic LDS pointer: low 32 bits are the LDS byte offset (ISA aperture rule)
  __builtin_amdgcn_global_load_async_to_lds_b128(
      (as1_v4i*)(uintptr_t)g, (as3_v4i*)(uint32_t)(uintptr_t)l, 0, 0);
#else
  *reinterpret_cast<v8h*>(l) = *reinterpret_cast<const v8h*>(g);
#endif
}

__device__ inline void wait_async_lds() {
#if HAVE_ASYNC
#if __has_builtin(__builtin_amdgcn_s_wait_asynccnt)
  __builtin_amdgcn_s_wait_asynccnt(0);
#else
  asm volatile("s_wait_asynccnt 0x0" ::: "memory");
#endif
#endif
}

// 16-half WMMA fragment from LDS (two aligned 16B ds reads)
__device__ inline v16h frag_lds(const _Float16* p) {
  v8h lo = *reinterpret_cast<const v8h*>(p);
  v8h hi = *reinterpret_cast<const v8h*>(p + 16);
  return __builtin_shufflevector(lo, hi, 0,1,2,3,4,5,6,7,8,9,10,11,12,13,14,15);
}

__device__ inline float gelu_tanh(float v) {
  float u = 0.7978845608028654f * (v + 0.044715f * v * v * v);
  return 0.5f * v * (1.0f + tanhf(u));
}

#define WMMA_F16(A_, B_, C_) \
  __builtin_amdgcn_wmma_f32_16x16x32_f16(false, (A_), false, (B_), (short)0, (C_), false, false)

// ---------------------------------------------------------------------------
// LDS double-buffered WMMA GEMM: C[M,N] = epi( A[M,K] * Bt[N,K]^T )
// 256 threads = 8 waves; tile 128(M) x 64(N) x 64(K); wave owns a 16x64 strip
// (4 accumulators, 8 v_wmma per tile). A/B tiles staged through LDS by the
// async engine (ASYNCcnt) with a 16B row pad: row stride 144B = 36 dwords;
// gcd(36,64)=4 => the 16 fragment-read lanes form a full bank permutation.
// All 10 fragment ds_loads for a tile are issued before the 8-WMMA chain so
// the waitcnt inserter can stage partial DScnt waits instead of draining.
// EPI: 0 scale*acc+bias->f32 | 1 gelu(acc+bias)->f16 | 2 acc+bias+addend->f32
//      3 Cf += acc
// ---------------------------------------------------------------------------
template <int EPI>
__global__ __launch_bounds__(256) void gemm_wmma_f16(
    const _Float16* __restrict__ A, const _Float16* __restrict__ Bt,
    const float* __restrict__ bias, const float* __restrict__ addend,
    float* __restrict__ Cf, _Float16* __restrict__ Ch,
    int M, int N, int K, float scale) {
  constexpr int BM = 128, BN = 64, BK = 64;
  constexpr int LDA = BK + 8;               // 72 halves = 144B row stride
  __shared__ _Float16 lds[2][BM * LDA + BN * LDA];

  const int tid  = threadIdx.x;
  const int wave = tid >> 5;
  const int lane = tid & 31;
  const int g    = lane >> 4;
  const int ln16 = lane & 15;
  const int m0   = blockIdx.y * BM;
  const int n0   = blockIdx.x * BN;

  auto copy_tile = [&](int k0, int buf) {
    _Float16* La = &lds[buf][0];
    _Float16* Lb = &lds[buf][BM * LDA];
    // A tile: 128 rows x 128B = 1024 chunks of 16B
#pragma unroll
    for (int i = 0; i < 4; ++i) {
      int c = tid + i * 256;
      int row = c >> 3, q = c & 7;
      int mrow = m0 + row; if (mrow >= M) mrow = M - 1;   // clamped duplicate
      cp16(A + (size_t)mrow * K + k0 + q * 8, La + row * LDA + q * 8);
    }
    // B tile: 64 rows x 128B = 512 chunks of 16B
#pragma unroll
    for (int i = 0; i < 2; ++i) {
      int c = tid + i * 256;
      int row = c >> 3, q = c & 7;
      cp16(Bt + (size_t)(n0 + row) * K + k0 + q * 8, Lb + row * LDA + q * 8);
    }
  };

  v8f acc0 = {}, acc1 = {}, acc2 = {}, acc3 = {};

  copy_tile(0, 0);
  wait_async_lds();
  __syncthreads();

  const int nk = K / BK;
  for (int kt = 0; kt < nk; ++kt) {
    const int cur = kt & 1;
    const _Float16* La = &lds[cur][(wave * 16 + ln16) * LDA + g * 8];
    const _Float16* Lb = &lds[cur][BM * LDA + ln16 * LDA + g * 8];

    // issue ALL fragment loads for this 64-deep tile first (20x ds_load_b128)
    v16h a0  = frag_lds(La);
    v16h a1  = frag_lds(La + 32);
    v16h b00 = frag_lds(Lb + 0 * 16 * LDA);
    v16h b10 = frag_lds(Lb + 1 * 16 * LDA);
    v16h b20 = frag_lds(Lb + 2 * 16 * LDA);
    v16h b30 = frag_lds(Lb + 3 * 16 * LDA);
    v16h b01 = frag_lds(Lb + 0 * 16 * LDA + 32);
    v16h b11 = frag_lds(Lb + 1 * 16 * LDA + 32);
    v16h b21 = frag_lds(Lb + 2 * 16 * LDA + 32);
    v16h b31 = frag_lds(Lb + 3 * 16 * LDA + 32);

    // kick the next tile's async copies while ds loads are in flight
    if (kt + 1 < nk) copy_tile((kt + 1) * BK, cur ^ 1);

    // 8 WMMAs in two 4-wide independent accumulator groups
    acc0 = WMMA_F16(a0, b00, acc0);
    acc1 = WMMA_F16(a0, b10, acc1);
    acc2 = WMMA_F16(a0, b20, acc2);
    acc3 = WMMA_F16(a0, b30, acc3);
    acc0 = WMMA_F16(a1, b01, acc0);
    acc1 = WMMA_F16(a1, b11, acc1);
    acc2 = WMMA_F16(a1, b21, acc2);
    acc3 = WMMA_F16(a1, b31, acc3);

    wait_async_lds();     // copies into buf^1 landed before anyone reuses it
    __syncthreads();
  }

  const int mw = m0 + wave * 16;
  v8f accs[4] = {acc0, acc1, acc2, acc3};
#pragma unroll
  for (int j = 0; j < 4; ++j) {
    int n = n0 + j * 16 + ln16;            // N is always a multiple of 64 here
    float bn = bias ? bias[n] : 0.0f;
#pragma unroll
    for (int r = 0; r < 8; ++r) {
      int m = mw + r + 8 * g;              // C/D layout: VGPR r -> row r + 8*g
      if (m >= M) continue;
      float v = accs[j][r] * scale + bn;
      size_t idx = (size_t)m * N + n;
      if (EPI == 0)      Cf[idx] = v;
      else if (EPI == 1) Ch[idx] = (_Float16)gelu_tanh(v);
      else if (EPI == 2) Cf[idx] = v + addend[idx];
      else               Cf[idx] += v;
    }
  }
}

// ---------------------------------------------------------------------------
// Weight prep: dst[n*Kp + k] = k<K ? (f16)src[k*N + n] : 0  (transpose+pad)
// ---------------------------------------------------------------------------
__global__ __launch_bounds__(256) void transpose_to_f16(
    const float* __restrict__ src, _Float16* __restrict__ dst,
    int K, int Kp, int N) {
  int t = blockIdx.x * blockDim.x + threadIdx.x;
  if (t >= Kp * N) return;
  int n = t / Kp, k = t - n * Kp;
  dst[(size_t)n * Kp + k] =
      (k < K) ? (_Float16)src[(size_t)k * N + n] : (_Float16)0.0f;
}

__global__ void combine_bias(const float* a, const float* b, float* out, int n) {
  int t = blockIdx.x * blockDim.x + threadIdx.x;
  if (t < n) out[t] = a[t] + b[t];
}

// zero h16[:, 544:576] pad columns (ws is poisoned, not re-zeroed, each launch)
__global__ __launch_bounds__(256) void zero_h_pad(_Float16* __restrict__ h16) {
  int t = blockIdx.x * blockDim.x + threadIdx.x;
  if (t >= N_NODES * (D_HP - D_H)) return;
  int n = t / (D_HP - D_H), f = t - n * (D_HP - D_H);
  h16[(size_t)n * D_HP + D_H + f] = (_Float16)0.0f;
}

// ---------------------------------------------------------------------------
// Dual LayerNorm (shared mean/var): writes z_att f16 (ld 512), h[:,0:512] f16
// ---------------------------------------------------------------------------
__global__ __launch_bounds__(256) void ln_dual(
    const float* __restrict__ x,
    const float* __restrict__ g1, const float* __restrict__ b1,
    const float* __restrict__ g2, const float* __restrict__ b2,
    _Float16* __restrict__ zatt, _Float16* __restrict__ h16) {
  __shared__ float sbuf[256];
  const int n = blockIdx.x;
  const float* xr = x + (size_t)n * D_EMB;
  const int i0 = threadIdx.x, i1 = threadIdx.x + 256;
  float v0 = xr[i0], v1 = xr[i1];

  sbuf[threadIdx.x] = v0 + v1;
  __syncthreads();
  for (int o = 128; o > 0; o >>= 1) {
    if (threadIdx.x < o) sbuf[threadIdx.x] += sbuf[threadIdx.x + o];
    __syncthreads();
  }
  float mu = sbuf[0] * (1.0f / D_EMB);
  __syncthreads();
  float d0 = v0 - mu, d1 = v1 - mu;
  sbuf[threadIdx.x] = d0 * d0 + d1 * d1;
  __syncthreads();
  for (int o = 128; o > 0; o >>= 1) {
    if (threadIdx.x < o) sbuf[threadIdx.x] += sbuf[threadIdx.x + o];
    __syncthreads();
  }
  float rstd = rsqrtf(sbuf[0] * (1.0f / D_EMB) + EPS_LN);
  float z0 = d0 * rstd, z1 = d1 * rstd;
  zatt[(size_t)n * D_EMB + i0] = (_Float16)(z0 * g1[i0] + b1[i0]);
  zatt[(size_t)n * D_EMB + i1] = (_Float16)(z1 * g1[i1] + b1[i1]);
  h16[(size_t)n * D_HP + i0] = (_Float16)(z0 * g2[i0] + b2[i0]);
  h16[(size_t)n * D_HP + i1] = (_Float16)(z1 * g2[i1] + b2[i1]);
}

// ---------------------------------------------------------------------------
// Per-(edge, head) q.k dot product (64-dim, float4 loads; lives in L2)
// ---------------------------------------------------------------------------
__global__ __launch_bounds__(256) void edge_logits(
    const float* __restrict__ q, const float* __restrict__ k,
    const int* __restrict__ ei, float* __restrict__ logits) {
  int t = blockIdx.x * blockDim.x + threadIdx.x;
  if (t >= N_EDGE * HEADS) return;
  int e = t >> 3, h = t & 7;
  int i = ei[e], j = ei[N_EDGE + e];
  const float4* qp = (const float4*)(q + (size_t)i * D_EMB + h * 64);
  const float4* kp = (const float4*)(k + (size_t)j * D_EMB + h * 64);
  float s = 0.0f;
#pragma unroll 4
  for (int d = 0; d < 16; ++d) {
    float4 a = qp[d], b = kp[d];
    s += a.x * b.x + a.y * b.y + a.z * b.z + a.w * b.w;
  }
  logits[(size_t)h * N_EDGE + e] = s;
}

__global__ __launch_bounds__(256) void zero_f32(float* p, int n) {
  int t = blockIdx.x * blockDim.x + threadIdx.x;
  if (t < n) p[t] = 0.0f;
}

// gather logits via edge_to_src, + bias, sigmoid, /dist, scatter-add feat
__global__ __launch_bounds__(256) void edge_scatter(
    const float* __restrict__ logits, const int* __restrict__ e2s,
    const float* __restrict__ att_bias, const float* __restrict__ dist,
    const float* __restrict__ src_pos, const int* __restrict__ src_index,
    float* __restrict__ feat) {
  int e = blockIdx.x * blockDim.x + threadIdx.x;
  if (e >= N_EDGE) return;
  int se = e2s[e];
  float d = dist[e];
  float inv = (d == 0.0f) ? 0.0f : 1.0f / d;   // att/inf == 0
  int r = src_index[e], c = src_index[N_EDGE + e];
  float vx = src_pos[c * 3 + 0], vy = src_pos[c * 3 + 1], vz = src_pos[c * 3 + 2];
  float* fr = feat + (size_t)r * FEAT;
#pragma unroll
  for (int h = 0; h < HEADS; ++h) {
    float l = logits[(size_t)h * N_EDGE + se] + att_bias[(size_t)h * N_EDGE + e];
    float a = inv / (1.0f + __expf(-l));
    atomicAdd(fr + h * 4 + 0, a * vx);
    atomicAdd(fr + h * 4 + 1, a * vy);
    atomicAdd(fr + h * 4 + 2, a * vz);
    atomicAdd(fr + h * 4 + 3, a);
  }
}

// feat[n,h,0:3] -= feat[n,h,3] * pos[n,:]
__global__ __launch_bounds__(256) void feat_pos(
    float* __restrict__ feat, const float* __restrict__ pos) {
  int t = blockIdx.x * blockDim.x + threadIdx.x;
  if (t >= N_NODES * HEADS) return;
  int n = t >> 3, h = t & 7;
  float w = feat[(size_t)n * FEAT + h * 4 + 3];
#pragma unroll
  for (int c = 0; c < 3; ++c)
    feat[(size_t)n * FEAT + h * 4 + c] -= w * pos[n * 3 + c];
}

// one block per feature: batch mean / rstd over N rows
__global__ __launch_bounds__(256) void bn_stats(
    const float* __restrict__ feat, float* __restrict__ stats) {
  __shared__ float s1[256], s2[256];
  int f = blockIdx.x;
  float a = 0.0f, b = 0.0f;
  for (int n = threadIdx.x; n < N_NODES; n += 256) {
    float v = feat[(size_t)n * FEAT + f];
    a += v; b += v * v;
  }
  s1[threadIdx.x] = a; s2[threadIdx.x] = b;
  __syncthreads();
  for (int o = 128; o > 0; o >>= 1) {
    if (threadIdx.x < o) { s1[threadIdx.x] += s1[threadIdx.x + o];
                           s2[threadIdx.x] += s2[threadIdx.x + o]; }
    __syncthreads();
  }
  if (threadIdx.x == 0) {
    float mu = s1[0] * (1.0f / N_NODES);
    float var = s2[0] * (1.0f / N_NODES) - mu * mu;
    stats[f] = mu;
    stats[FEAT + f] = rsqrtf(var + EPS_LN);
  }
}

// normalized feat -> h[:, 512:544] as f16
__global__ __launch_bounds__(256) void bn_apply(
    const float* __restrict__ feat, const float* __restrict__ stats,
    const float* __restrict__ bng, const float* __restrict__ bnb,
    _Float16* __restrict__ h16) {
  int t = blockIdx.x * blockDim.x + threadIdx.x;
  if (t >= N_NODES * FEAT) return;
  int n = t >> 5, f = t & 31;
  float v = (feat[t] - stats[f]) * stats[FEAT + f] * bng[f] + bnb[f];
  h16[(size_t)n * D_HP + D_EMB + f] = (_Float16)v;
}

// ---------------------------------------------------------------------------
extern "C" void kernel_launch(void* const* d_in, const int* in_sizes, int n_in,
                              void* d_out, int out_size, void* d_ws, size_t ws_size,
                              hipStream_t stream) {
  const float* x        = (const float*)d_in[0];
  const int*   eidx     = (const int*)d_in[1];
  const int*   sidx     = (const int*)d_in[2];
  const int*   e2s      = (const int*)d_in[3];
  const float* att_bias = (const float*)d_in[4];
  const float* dist     = (const float*)d_in[5];
  const float* pos      = (const float*)d_in[6];
  const float* src_pos  = (const float*)d_in[7];
  const float* Wq = (const float*)d_in[8];  const float* bq = (const float*)d_in[9];
  const float* Wk = (const float*)d_in[10]; const float* bk = (const float*)d_in[11];
  const float* lag = (const float*)d_in[12]; const float* lab = (const float*)d_in[13];
  const float* lmg = (const float*)d_in[14]; const float* lmb = (const float*)d_in[15];
  const float* bng = (const float*)d_in[16]; const float* bnb = (const float*)d_in[17];
  const float* W1 = (const float*)d_in[18]; const float* b1 = (const float*)d_in[19];
  const float* W2 = (const float*)d_in[20]; const float* b2 = (const float*)d_in[21];
  const float* Ws = (const float*)d_in[22]; const float* bs = (const float*)d_in[23];
  float* out = (float*)d_out;

  // -------- carve scratch (256B aligned) --------
  char* base = (char*)d_ws;
  size_t off = 0;
  auto carve = [&](size_t bytes) -> char* {
    char* p = base + off;
    off = (off + bytes + 255) & ~(size_t)255;
    return p;
  };
  _Float16* zatt  = (_Float16*)carve((size_t)N_NODES * D_EMB * 2);
  _Float16* h16   = (_Float16*)carve((size_t)N_NODES * D_HP * 2);
  _Float16* hid16 = (_Float16*)carve((size_t)N_NODES * D_HID * 2);
  float*    qf    = (float*)carve((size_t)N_NODES * D_EMB * 4);
  float*    kf    = (float*)carve((size_t)N_NODES * D_EMB * 4);
  float*    lg    = (float*)carve((size_t)HEADS * N_EDGE * 4);
  float*    feat  = (float*)carve((size_t)N_NODES * FEAT * 4);
  float*    stats = (float*)carve(2 * FEAT * 4);
  float*    bcomb = (float*)carve(D_EMB * 4);
  _Float16* WqT = (_Float16*)carve((size_t)D_EMB * D_EMB * 2);
  _Float16* WkT = (_Float16*)carve((size_t)D_EMB * D_EMB * 2);
  _Float16* W1T = (_Float16*)carve((size_t)D_HID * D_HP * 2);
  _Float16* W2T = (_Float16*)carve((size_t)D_EMB * D_HID * 2);
  _Float16* WsT = (_Float16*)carve((size_t)D_EMB * D_HP * 2);
  (void)ws_size; (void)n_in; (void)in_sizes; (void)out_size;

  auto cdiv = [](int a, int b) { return (a + b - 1) / b; };

  // -------- weight prep: transpose + f32->f16 (+ K padding to 576) --------
  transpose_to_f16<<<cdiv(D_EMB * D_EMB, 256), 256, 0, stream>>>(Wq, WqT, D_EMB, D_EMB, D_EMB);
  transpose_to_f16<<<cdiv(D_EMB * D_EMB, 256), 256, 0, stream>>>(Wk, WkT, D_EMB, D_EMB, D_EMB);
  transpose_to_f16<<<cdiv(D_HP * D_HID, 256), 256, 0, stream>>>(W1, W1T, D_H, D_HP, D_HID);
  transpose_to_f16<<<cdiv(D_HID * D_EMB, 256), 256, 0, stream>>>(W2, W2T, D_HID, D_HID, D_EMB);
  transpose_to_f16<<<cdiv(D_HP * D_EMB, 256), 256, 0, stream>>>(Ws, WsT, D_H, D_HP, D_EMB);
  combine_bias<<<cdiv(D_EMB, 256), 256, 0, stream>>>(bs, b2, bcomb, D_EMB);
  zero_h_pad<<<cdiv(N_NODES * (D_HP - D_H), 256), 256, 0, stream>>>(h16);

  // -------- LN (shared stats) -> z_att f16, h[:,0:512] f16 --------
  ln_dual<<<N_NODES, 256, 0, stream>>>(x, lag, lab, lmg, lmb, zatt, h16);

  // -------- q/k projections (WMMA) --------
  dim3 g_qk(D_EMB / 64, cdiv(N_NODES, 128));
  gemm_wmma_f16<0><<<g_qk, 256, 0, stream>>>(zatt, WqT, bq, nullptr, qf, nullptr,
                                             N_NODES, D_EMB, D_EMB, 0.125f);
  gemm_wmma_f16<0><<<g_qk, 256, 0, stream>>>(zatt, WkT, bk, nullptr, kf, nullptr,
                                             N_NODES, D_EMB, D_EMB, 1.0f);

  // -------- edge attention + scatter --------
  zero_f32<<<cdiv(N_NODES * FEAT, 256), 256, 0, stream>>>(feat, N_NODES * FEAT);
  edge_logits<<<cdiv(N_EDGE * HEADS, 256), 256, 0, stream>>>(qf, kf, eidx, lg);
  edge_scatter<<<cdiv(N_EDGE, 256), 256, 0, stream>>>(lg, e2s, att_bias, dist,
                                                      src_pos, sidx, feat);
  feat_pos<<<cdiv(N_NODES * HEADS, 256), 256, 0, stream>>>(feat, pos);
  bn_stats<<<FEAT, 256, 0, stream>>>(feat, stats);
  bn_apply<<<cdiv(N_NODES * FEAT, 256), 256, 0, stream>>>(feat, stats, bng, bnb, h16);

  // -------- MLP (WMMA) --------
  dim3 g_h1(D_HID / 64, cdiv(N_NODES, 128));
  gemm_wmma_f16<1><<<g_h1, 256, 0, stream>>>(h16, W1T, b1, nullptr, nullptr, hid16,
                                             N_NODES, D_HID, D_HP, 1.0f);
  dim3 g_o(D_EMB / 64, cdiv(N_NODES, 128));
  // out = h@Ws + (bs + b2) + x
  gemm_wmma_f16<2><<<g_o, 256, 0, stream>>>(h16, WsT, bcomb, x, out, nullptr,
                                            N_NODES, D_EMB, D_HP, 1.0f);
  // out += hidden@W2
  gemm_wmma_f16<3><<<g_o, 256, 0, stream>>>(hid16, W2T, nullptr, nullptr, out, nullptr,
                                            N_NODES, D_EMB, D_HID, 1.0f);
}
